// LSTM_33432025432443
// MI455X (gfx1250) — matrix operand order repaired and checked by reading, not compiled, not run on hardware
//
#include <hip/hip_runtime.h>
#include <hip/hip_bf16.h>

typedef __attribute__((ext_vector_type(16))) _Float16 v16h;
typedef __attribute__((ext_vector_type(8)))  _Float16 v8h;
typedef __attribute__((ext_vector_type(8)))  float    v8f;

#define H    64
#define G    256   // 4*H
#define T    512
#define ROWS 16    // sequences per workgroup

#define LOG2E 1.4426950408889634f

__device__ __forceinline__ float fast_rcp(float x) {
    return __builtin_amdgcn_rcpf(x);            // v_rcp_f32 (fast, no IEEE fixup)
}
__device__ __forceinline__ float fast_exp2(float x) {
    return __builtin_amdgcn_exp2f(x);           // v_exp_f32 directly (base-2)
}
// tanh(x) = 2*sigmoid(2x) - 1 = 2/(1+exp2(-2*log2e*x)) - 1, all fast ops,
// single pre-exp multiply (folded constant).
__device__ __forceinline__ float fast_tanh(float x) {
    return __builtin_fmaf(2.0f, fast_rcp(1.0f + fast_exp2(-2.0f * LOG2E * x)), -1.0f);
}

// B fragment for v_wmma_f32_16x16x32_f16.
// B is W^T (64x256): B[k][n] = W[n][k], W row-major (256,64) fp16.
// Lane layout: column n = gcol (= tile*16 + lane%16), K = 32*kk + (lane/16)*16 + {0..15},
// i.e. 16 contiguous halfs of row n of W.
__device__ __forceinline__ v16h loadB(const _Float16* W, int gcol, int laneHi, int kk) {
    return *(const v16h*)(W + gcol * H + kk * 32 + laneHi * 16);
}

// A fragment from LDS fp16 h-tile (16 x 64, row-major).
// A 16x32 layout (matches loadB): lane row m = lane%16, half = lane/16:
//  elems 0..15 -> K = 32*kk + half*16 + {0..15}
__device__ __forceinline__ v16h loadA(const _Float16* hbuf, int lane, int kk) {
    const int m  = lane & 15;
    const int hs = (lane >> 4) * 16;
    const int base = m * H + kk * 32 + hs;
    v8h lo = *(const v8h*)(hbuf + base);
    v8h hi = *(const v8h*)(hbuf + base + 8);
    v16h a;
#pragma unroll
    for (int i = 0; i < 8; ++i) { a[i] = lo[i]; a[i + 8] = hi[i]; }
    return a;
}

__global__ void lstm_convert_weights(const float* __restrict__ whh1,
                                     const float* __restrict__ wih2,
                                     const float* __restrict__ whh2,
                                     _Float16* __restrict__ ws) {
    int i = blockIdx.x * blockDim.x + threadIdx.x;
    if (i < G * H) {
        ws[i]             = (_Float16)whh1[i];
        ws[G * H + i]     = (_Float16)wih2[i];
        ws[2 * G * H + i] = (_Float16)whh2[i];
    }
}

__global__ __launch_bounds__(512) void lstm_seq_kernel(
    const float* __restrict__ y,        // (N, T)
    const float* __restrict__ W_ih1,    // (256, 1)
    const float* __restrict__ b_ih1, const float* __restrict__ b_hh1,
    const float* __restrict__ b_ih2, const float* __restrict__ b_hh2,
    const float* __restrict__ W_lin,    // (1, 64)
    const float* __restrict__ b_lin,
    const _Float16* __restrict__ ws,    // fp16 weights: whh1 | wih2 | whh2
    float* __restrict__ out,
    int outw, int skip)
{
    __shared__ float    x_sh[ROWS];
    __shared__ _Float16 h1s[ROWS * H];
    __shared__ _Float16 h2s[ROWS * H];
    __shared__ float    gact[ROWS * G];   // activated gates of current layer

    const int tid    = threadIdx.x;
    const int wave   = tid >> 5;          // 0..15 : gate column tile
    const int lane   = tid & 31;
    const int laneH  = lane & 15;
    const int laneHi = lane >> 4;
    const int n0     = blockIdx.x * ROWS;

    const _Float16* Whh1h = ws;
    const _Float16* Wih2h = ws + G * H;
    const _Float16* Whh2h = ws + 2 * G * H;

    // ---- per-lane invariants ----
    const int gcol = wave * 16 + laneH;                 // gate column 0..255
    const bool isG = ((wave >> 2) == 2);                // g-gate uses tanh
    // branch-free activation: act = A * rcp(1 + exp2(SL2E * x)) + B
    //   sigmoid: SL2E=-log2e,   A=1, B=0
    //   tanh:    SL2E=-2*log2e, A=2, B=-1   (tanh(x)=2*sigmoid(2x)-1)
    const float actS = isG ? (-2.0f * LOG2E) : (-LOG2E);
    const float actA = isG ?  2.0f :  1.0f;
    const float actB = isG ? -1.0f :  0.0f;

    const float bias1 = b_ih1[gcol] + b_hh1[gcol];
    const float bias2 = b_ih2[gcol] + b_hh2[gcol];
    const float wx    = W_ih1[gcol];                    // D == 1
    const float wlin_a = W_lin[lane];
    const float wlin_b = W_lin[lane + 32];
    const float blin   = b_lin[0];

    // B fragments held in registers for the whole sequence
    const v16h bhh1_0 = loadB(Whh1h, gcol, laneHi, 0);
    const v16h bhh1_1 = loadB(Whh1h, gcol, laneHi, 1);
    const v16h bih2_0 = loadB(Wih2h, gcol, laneHi, 0);
    const v16h bih2_1 = loadB(Wih2h, gcol, laneHi, 1);
    const v16h bhh2_0 = loadB(Whh2h, gcol, laneHi, 0);
    const v16h bhh2_1 = loadB(Whh2h, gcol, laneHi, 1);

    // per-thread cell state: elements e0 = 2*tid, e0+1 of the 16x64 tile
    const int e0   = tid * 2;
    const int urow = e0 >> 6;
    const int uj   = e0 & 63;
    float c1a = 0.f, c1b = 0.f, c2a = 0.f, c2b = 0.f;

    // zero h state
    h1s[e0] = (_Float16)0.f; h1s[e0 + 1] = (_Float16)0.f;
    h2s[e0] = (_Float16)0.f; h2s[e0 + 1] = (_Float16)0.f;
    __syncthreads();

    for (int t = 0; t < T; ++t) {
        if (tid < ROWS) x_sh[tid] = y[(size_t)(n0 + tid) * T + t];
        __syncthreads();                           // B1: x ready, h1s/h2s from prev step ready

        // ---------- layer 1 gates: bias + x*W_ih1 + h1 @ W_hh1^T ----------
        v8f acc;
#pragma unroll
        for (int r = 0; r < 8; ++r)
            acc[r] = __builtin_fmaf(x_sh[r + 8 * laneHi], wx, bias1);

        {
            v16h a0 = loadA(h1s, lane, 0);
            acc = __builtin_amdgcn_wmma_f32_16x16x32_f16(false, a0, false, bhh1_0,
                                                         (short)0, acc, false, false);
            v16h a1 = loadA(h1s, lane, 1);
            acc = __builtin_amdgcn_wmma_f32_16x16x32_f16(false, a1, false, bhh1_1,
                                                         (short)0, acc, false, false);
        }
#pragma unroll
        for (int r = 0; r < 8; ++r) {
            float av = __builtin_fmaf(
                actA, fast_rcp(1.0f + fast_exp2(actS * acc[r])), actB);
            gact[(r + 8 * laneHi) * G + gcol] = av;
        }
        __syncthreads();                           // B2: layer-1 gates ready

        // ---------- c1 / h1 update (2 elements per thread) ----------
        {
            const float* gr = gact + urow * G;
            float i0 = gr[uj],     f0 = gr[64 + uj],     g0 = gr[128 + uj],     o0 = gr[192 + uj];
            float i1 = gr[uj + 1], f1 = gr[64 + uj + 1], g1 = gr[128 + uj + 1], o1 = gr[192 + uj + 1];
            c1a = __builtin_fmaf(f0, c1a, i0 * g0);
            c1b = __builtin_fmaf(f1, c1b, i1 * g1);
            h1s[urow * H + uj]     = (_Float16)(o0 * fast_tanh(c1a));
            h1s[urow * H + uj + 1] = (_Float16)(o1 * fast_tanh(c1b));
        }
        __syncthreads();                           // B3: h1 ready, gact free

        // ---------- layer 2 gates: bias + h1 @ W_ih2^T + h2 @ W_hh2^T ----------
#pragma unroll
        for (int r = 0; r < 8; ++r) acc[r] = bias2;
        {
            v16h a0 = loadA(h1s, lane, 0);
            acc = __builtin_amdgcn_wmma_f32_16x16x32_f16(false, a0, false, bih2_0,
                                                         (short)0, acc, false, false);
            v16h a1 = loadA(h1s, lane, 1);
            acc = __builtin_amdgcn_wmma_f32_16x16x32_f16(false, a1, false, bih2_1,
                                                         (short)0, acc, false, false);
            v16h a2 = loadA(h2s, lane, 0);
            acc = __builtin_amdgcn_wmma_f32_16x16x32_f16(false, a2, false, bhh2_0,
                                                         (short)0, acc, false, false);
            v16h a3 = loadA(h2s, lane, 1);
            acc = __builtin_amdgcn_wmma_f32_16x16x32_f16(false, a3, false, bhh2_1,
                                                         (short)0, acc, false, false);
        }
#pragma unroll
        for (int r = 0; r < 8; ++r) {
            float av = __builtin_fmaf(
                actA, fast_rcp(1.0f + fast_exp2(actS * acc[r])), actB);
            gact[(r + 8 * laneHi) * G + gcol] = av;
        }
        __syncthreads();                           // B4: layer-2 gates ready

        // ---------- c2 / h2 update ----------
        {
            const float* gr = gact + urow * G;
            float i0 = gr[uj],     f0 = gr[64 + uj],     g0 = gr[128 + uj],     o0 = gr[192 + uj];
            float i1 = gr[uj + 1], f1 = gr[64 + uj + 1], g1 = gr[128 + uj + 1], o1 = gr[192 + uj + 1];
            c2a = __builtin_fmaf(f0, c2a, i0 * g0);
            c2b = __builtin_fmaf(f1, c2b, i1 * g1);
            h2s[urow * H + uj]     = (_Float16)(o0 * fast_tanh(c2a));
            h2s[urow * H + uj + 1] = (_Float16)(o1 * fast_tanh(c2b));
        }
        __syncthreads();                           // B5: h2 ready

        // ---------- output projection: wave w -> row w ----------
        if (t >= skip) {
            const int row = wave;
            float p = (float)h2s[row * H + lane] * wlin_a
                    + (float)h2s[row * H + lane + 32] * wlin_b;
#pragma unroll
            for (int off = 16; off > 0; off >>= 1)
                p += __shfl_xor(p, off, 32);
            if (lane == 0)
                out[(size_t)(n0 + row) * outw + (t - skip)] = p + blin;
        }
        // reads of h2s are protected by B1..B4 of the next iteration
    }
}

extern "C" void kernel_launch(void* const* d_in, const int* in_sizes, int n_in,
                              void* d_out, int out_size, void* d_ws, size_t ws_size,
                              hipStream_t stream) {
    const float* y     = (const float*)d_in[0];
    const float* W_ih1 = (const float*)d_in[1];
    const float* W_hh1 = (const float*)d_in[2];
    const float* b_ih1 = (const float*)d_in[3];
    const float* b_hh1 = (const float*)d_in[4];
    const float* W_ih2 = (const float*)d_in[5];
    const float* W_hh2 = (const float*)d_in[6];
    const float* b_ih2 = (const float*)d_in[7];
    const float* b_hh2 = (const float*)d_in[8];
    const float* W_lin = (const float*)d_in[9];
    const float* b_lin = (const float*)d_in[10];

    const int N    = in_sizes[0] / T;       // 2048
    const int outw = out_size / N;          // 511
    const int skip = T - outw;              // pre_output_len == 1

    _Float16* wsh = (_Float16*)d_ws;        // 3 * 256 * 64 fp16 = 96 KB

    lstm_convert_weights<<<(G * H + 255) / 256, 256, 0, stream>>>(W_hh1, W_ih2, W_hh2, wsh);

    lstm_seq_kernel<<<N / ROWS, 512, 0, stream>>>(
        y, W_ih1, b_ih1, b_hh1, b_ih2, b_hh2, W_lin, b_lin,
        wsh, (float*)d_out, outw, skip);
}
